// decoder_mha_81741817578304
// MI455X (gfx1250) — matrix operand order repaired and checked by reading, not compiled, not run on hardware
//
#include <hip/hip_runtime.h>
#include <hip/hip_bf16.h>

typedef __bf16 bf16_t;
typedef __attribute__((ext_vector_type(16))) __bf16 v16bf;
typedef __attribute__((ext_vector_type(8)))  float  v8f;

#define B_  2
#define S_  2048
#define D_  1024
#define H_  16
#define KD_ 64

// ---------------------------------------------------------------------------
// Fragment loaders (layouts per CDNA5 ISA 7.12.2, wave32)
// A 16x32 bf16: lane holds row = lane%16; elements 0..7 = K kb..kb+7,
//               elements 8..15 = K kb+16..kb+23, kb = 8*(lane/16)
// B 32x16 bf16: lane holds col n = lane%16; elements 0..7 = K kb..kb+7,
//               elements 8..15 = K kb+16..kb+23, kb = 8*(lane/16). Memory holds B^T rows (n-major).
// C 16x16 f32 : element i -> row m = i + 8*(lane/16), col n = lane%16
// ---------------------------------------------------------------------------
__device__ __forceinline__ v16bf load_a_frag(const bf16_t* __restrict__ base,
                                             int ld, int row0, int d0, int lane) {
    int m  = row0 + (lane & 15);
    int kb = d0 + ((lane >> 4) << 3);
    const bf16_t* p = base + (size_t)m * ld + kb;
    union { uint4 u[2]; v16bf v; } t;
    t.u[0] = *(const uint4*)(p);
    t.u[1] = *(const uint4*)(p + 16);
    return t.v;
}

__device__ __forceinline__ v16bf load_b_frag(const bf16_t* __restrict__ base,
                                             int ld, int n0, int d0, int lane) {
    int n  = n0 + (lane & 15);
    int kb = d0 + ((lane >> 4) << 3);
    const bf16_t* p = base + (size_t)n * ld + kb;
    union { uint4 u[2]; v16bf v; } t;
    t.u[0] = *(const uint4*)(p);
    t.u[1] = *(const uint4*)(p + 16);
    return t.v;
}

__device__ __forceinline__ v8f wmma_bf16(v16bf a, v16bf b, v8f c) {
    return __builtin_amdgcn_wmma_f32_16x16x32_bf16(
        /*neg_a=*/false, a, /*neg_b=*/false, b,
        /*c_mod=*/(short)0, c, /*reuse_a=*/false, /*reuse_b=*/false);
}

// ---------------------------------------------------------------------------
// fp32 -> bf16 convert
// ---------------------------------------------------------------------------
__global__ void cvt_f32_bf16(const float* __restrict__ in,
                             bf16_t* __restrict__ out, int n) {
    int i = blockIdx.x * blockDim.x + threadIdx.x;
    if (i < n) out[i] = (bf16_t)in[i];
}

// ---------------------------------------------------------------------------
// GEMM: Y[r,e] = sum_d A[r,d] * W[e,d]   (A: Mx1024 bf16, W: 1024x1024 bf16)
// One wave computes a 32x64 tile (2 row blocks x 4 col frags): each B frag
// feeds 2 WMMAs -> ~21 flop/byte from L2.
// mode 0: store bf16 to [B,H,S,KD]   (Q, K pre-smear)
// mode 1: store bf16 to [B,H,KD,S]   (V transposed)
// mode 2: store f32 row-major + bias (final output)
// ---------------------------------------------------------------------------
__global__ __launch_bounds__(32)
void gemm_bf16(const bf16_t* __restrict__ A, const bf16_t* __restrict__ W,
               void* __restrict__ out, const float* __restrict__ bias, int mode) {
    const int lane = threadIdx.x;
    const int r0 = blockIdx.x * 32;
    const int e0 = blockIdx.y * 64;

    v8f acc[2][4] = {};
    for (int d0 = 0; d0 < D_; d0 += 32) {
        v16bf a0 = load_a_frag(A, D_, r0,      d0, lane);
        v16bf a1 = load_a_frag(A, D_, r0 + 16, d0, lane);
#pragma unroll
        for (int f = 0; f < 4; ++f) {
            v16bf b = load_b_frag(W, D_, e0 + f * 16, d0, lane);
            acc[0][f] = wmma_bf16(a0, b, acc[0][f]);
            acc[1][f] = wmma_bf16(a1, b, acc[1][f]);
        }
    }

    const int hv = lane >> 4;
    const int n  = lane & 15;
#pragma unroll
    for (int rb = 0; rb < 2; ++rb) {
#pragma unroll
        for (int f = 0; f < 4; ++f) {
#pragma unroll
            for (int i = 0; i < 8; ++i) {
                int m = i + 8 * hv;
                int r = r0 + rb * 16 + m;       // 0..B*S-1
                int e = e0 + f * 16 + n;        // 0..D-1
                float v = acc[rb][f][i];
                if (mode == 2) {
                    ((float*)out)[(size_t)r * D_ + e] = v + bias[e];
                } else {
                    int bb = r >> 11;           // r / S_
                    int s  = r & (S_ - 1);
                    int h  = e >> 6;            // e / KD_
                    int kd = e & (KD_ - 1);
                    size_t idx = (mode == 0)
                        ? ((((size_t)bb * H_ + h) * S_ + s) * KD_ + kd)
                        : ((((size_t)bb * H_ + h) * KD_ + kd) * S_ + s);
                    ((bf16_t*)out)[idx] = (bf16_t)v;
                }
            }
        }
    }
}

// ---------------------------------------------------------------------------
// Key smear: k[s] = sig(alpha[h,s-1])*k[s] + (1-sig)*k[s-1], s>=1
// ---------------------------------------------------------------------------
__global__ void smear_k(const bf16_t* __restrict__ kin,
                        const float* __restrict__ alpha,
                        bf16_t* __restrict__ kout) {
    size_t i = (size_t)blockIdx.x * blockDim.x + threadIdx.x;
    const size_t total = (size_t)B_ * H_ * S_ * KD_;
    if (i >= total) return;
    size_t t = i / KD_;
    int s = (int)(t % S_);
    int h = (int)((t / S_) % H_);
    float cur = (float)kin[i];
    if (s == 0) { kout[i] = (bf16_t)cur; return; }
    float prev = (float)kin[i - KD_];
    float a  = alpha[h * (S_ - 1) + (s - 1)];
    float it = 1.0f / (1.0f + __expf(-a));
    kout[i] = (bf16_t)(cur * it + prev * (1.0f - it));
}

// ---------------------------------------------------------------------------
// Causal flash attention, one wave per (b, h, 32-query tile).
// Two 16-row blocks share every K and V fragment -> 16 WMMAs per 8KB tile.
// q,k: [B,H,S,KD] bf16; vT: [B,H,KD,S] bf16; mha out: [B,S,D] bf16
// ---------------------------------------------------------------------------
__global__ __launch_bounds__(32)
void attn_kernel(const bf16_t* __restrict__ q, const bf16_t* __restrict__ k,
                 const bf16_t* __restrict__ vT, bf16_t* __restrict__ mha) {
    __shared__ float pshm[32 * 32];

    const int lane = threadIdx.x;
    const int q0 = blockIdx.x * 32;
    const int h  = blockIdx.y;
    const int bb = blockIdx.z;

    const bf16_t* qh = q  + (size_t)(bb * H_ + h) * S_ * KD_;
    const bf16_t* kh = k  + (size_t)(bb * H_ + h) * S_ * KD_;
    const bf16_t* vh = vT + (size_t)(bb * H_ + h) * KD_ * S_;

    v16bf qa[2][2];
#pragma unroll
    for (int rb = 0; rb < 2; ++rb) {
        qa[rb][0] = load_a_frag(qh, KD_, q0 + rb * 16, 0,  lane);
        qa[rb][1] = load_a_frag(qh, KD_, q0 + rb * 16, 32, lane);
    }

    const int hv = lane >> 4;
    const int n  = lane & 15;
    const float scale = 0.125f;   // 1/sqrt(64)

    float mrow[2][8], lrow[2][8];
    v8f o[2][4] = {};
#pragma unroll
    for (int rb = 0; rb < 2; ++rb)
#pragma unroll
        for (int i = 0; i < 8; ++i) { mrow[rb][i] = -__builtin_inff(); lrow[rb][i] = 0.0f; }

    const int kend = q0 + 32;
    for (int k0 = 0; k0 < kend; k0 += 32) {
        // prefetch next tile of K and V into cache (global_prefetch_b8)
        if (k0 + 32 < kend) {
            __builtin_prefetch(kh + (size_t)(k0 + 32 + n) * KD_ + hv * 32, 0, 0);
            __builtin_prefetch(vh + (size_t)(lane * 2) * S_ + k0 + 32, 0, 0);
        }

        // ---- K tile fragments, shared by both row blocks ----
        v16bf kb00 = load_b_frag(kh, KD_, k0,      0,  lane);
        v16bf kb01 = load_b_frag(kh, KD_, k0,      32, lane);
        v16bf kb10 = load_b_frag(kh, KD_, k0 + 16, 0,  lane);
        v16bf kb11 = load_b_frag(kh, KD_, k0 + 16, 32, lane);

        __syncthreads();   // WAR: pshm reads of previous iteration are done
#pragma unroll
        for (int rb = 0; rb < 2; ++rb) {
            // ---- scores: Q (16x64) x K_tile^T (64x32) -> two 16x16 frags ----
            v8f s0 = {}, s1 = {};
            s0 = wmma_bf16(qa[rb][0], kb00, s0);
            s0 = wmma_bf16(qa[rb][1], kb01, s0);
            s1 = wmma_bf16(qa[rb][0], kb10, s1);
            s1 = wmma_bf16(qa[rb][1], kb11, s1);

            // ---- scale + causal mask ----
            float sm0[8], sm1[8];
#pragma unroll
            for (int i = 0; i < 8; ++i) {
                int m  = i + 8 * hv;
                int qg = q0 + rb * 16 + m;
                sm0[i] = ((k0 + n)      <= qg) ? s0[i] * scale : -__builtin_inff();
                sm1[i] = ((k0 + 16 + n) <= qg) ? s1[i] * scale : -__builtin_inff();
            }

            // ---- row max (reduce over 16 lanes of each half) ----
            float mx[8];
#pragma unroll
            for (int i = 0; i < 8; ++i) mx[i] = fmaxf(sm0[i], sm1[i]);
#pragma unroll
            for (int off = 1; off < 16; off <<= 1)
#pragma unroll
                for (int i = 0; i < 8; ++i)
                    mx[i] = fmaxf(mx[i], __shfl_xor(mx[i], off, 16));

            // ---- online softmax update ----
            float corr[8], p0[8], p1[8], ps[8];
#pragma unroll
            for (int i = 0; i < 8; ++i) {
                float mnew = fmaxf(mrow[rb][i], mx[i]);
                corr[i] = __expf(mrow[rb][i] - mnew);
                p0[i]   = __expf(sm0[i] - mnew);
                p1[i]   = __expf(sm1[i] - mnew);
                ps[i]   = p0[i] + p1[i];
                mrow[rb][i] = mnew;
            }
#pragma unroll
            for (int off = 1; off < 16; off <<= 1)
#pragma unroll
                for (int i = 0; i < 8; ++i)
                    ps[i] += __shfl_xor(ps[i], off, 16);
#pragma unroll
            for (int i = 0; i < 8; ++i) lrow[rb][i] = lrow[rb][i] * corr[i] + ps[i];
#pragma unroll
            for (int f = 0; f < 4; ++f)
#pragma unroll
                for (int i = 0; i < 8; ++i) o[rb][f][i] *= corr[i];

            // ---- stage P block (C layout) into LDS ----
#pragma unroll
            for (int i = 0; i < 8; ++i) {
                int mr = rb * 16 + i + 8 * hv;
                pshm[mr * 32 + n]      = p0[i];
                pshm[mr * 32 + 16 + n] = p1[i];
            }
        }
        __syncthreads();

        // ---- rebuild P as A-layout fragments (both row blocks) ----
        v16bf pv[2];
        {
            int row = lane & 15;
            int kb  = (lane >> 4) * 8;
#pragma unroll
            for (int rb = 0; rb < 2; ++rb) {
                const float* pr = &pshm[(rb * 16 + row) * 32];
#pragma unroll
                for (int j = 0; j < 8; ++j) {
                    pv[rb][j]     = (bf16_t)pr[kb + j];
                    pv[rb][8 + j] = (bf16_t)pr[kb + 16 + j];
                }
            }
        }

        // ---- O += P (16x32) x V_tile (32x64); each V frag feeds 2 WMMAs ----
#pragma unroll
        for (int f = 0; f < 4; ++f) {
            v16bf vb = load_b_frag(vh, S_, f * 16, k0, lane);
            o[0][f] = wmma_bf16(pv[0], vb, o[0][f]);
            o[1][f] = wmma_bf16(pv[1], vb, o[1][f]);
        }
    }

    // ---- normalize + store to [B,S,D] ----
#pragma unroll
    for (int rb = 0; rb < 2; ++rb) {
#pragma unroll
        for (int f = 0; f < 4; ++f) {
#pragma unroll
            for (int i = 0; i < 8; ++i) {
                int m  = i + 8 * hv;
                int s  = q0 + rb * 16 + m;
                int kd = f * 16 + n;
                float val = o[rb][f][i] / lrow[rb][i];
                mha[((size_t)bb * S_ + s) * D_ + h * KD_ + kd] = (bf16_t)val;
            }
        }
    }
}

// ---------------------------------------------------------------------------
// Host orchestration
// ---------------------------------------------------------------------------
extern "C" void kernel_launch(void* const* d_in, const int* in_sizes, int n_in,
                              void* d_out, int out_size, void* d_ws, size_t ws_size,
                              hipStream_t stream) {
    const float* x     = (const float*)d_in[0];
    const float* W_q   = (const float*)d_in[1];
    const float* W_k   = (const float*)d_in[2];
    const float* W_v   = (const float*)d_in[3];
    const float* W_o   = (const float*)d_in[4];
    const float* b_o   = (const float*)d_in[5];
    const float* alpha = (const float*)d_in[6];

    const size_t NX = (size_t)B_ * S_ * D_;        // 4,194,304
    const size_t NW = (size_t)D_ * D_;             // 1,048,576
    const size_t NH = (size_t)B_ * H_ * S_ * KD_;  // == NX

    bf16_t* ws = (bf16_t*)d_ws;
    bf16_t* xb  = ws;            ws += NX;
    bf16_t* wqb = ws;            ws += NW;
    bf16_t* wkb = ws;            ws += NW;
    bf16_t* wvb = ws;            ws += NW;
    bf16_t* wob = ws;            ws += NW;
    bf16_t* qb  = ws;            ws += NH;   // [B,H,S,KD]
    bf16_t* kb  = ws;            ws += NH;   // [B,H,S,KD] pre-smear
    bf16_t* ksb = ws;            ws += NH;   // [B,H,S,KD] smeared
    bf16_t* vtb = ws;            ws += NH;   // [B,H,KD,S]
    bf16_t* mhb = ws;            ws += NX;   // [B,S,D]

    // 1) fp32 -> bf16 converts
    cvt_f32_bf16<<<dim3((NX + 255) / 256), dim3(256), 0, stream>>>(x,   xb,  (int)NX);
    cvt_f32_bf16<<<dim3((NW + 255) / 256), dim3(256), 0, stream>>>(W_q, wqb, (int)NW);
    cvt_f32_bf16<<<dim3((NW + 255) / 256), dim3(256), 0, stream>>>(W_k, wkb, (int)NW);
    cvt_f32_bf16<<<dim3((NW + 255) / 256), dim3(256), 0, stream>>>(W_v, wvb, (int)NW);
    cvt_f32_bf16<<<dim3((NW + 255) / 256), dim3(256), 0, stream>>>(W_o, wob, (int)NW);

    // 2) projections: grid = (M/32, N/64)
    dim3 ggrid(B_ * S_ / 32, D_ / 64);
    gemm_bf16<<<ggrid, dim3(32), 0, stream>>>(xb, wqb, qb,  nullptr, 0);
    gemm_bf16<<<ggrid, dim3(32), 0, stream>>>(xb, wkb, kb,  nullptr, 0);
    gemm_bf16<<<ggrid, dim3(32), 0, stream>>>(xb, wvb, vtb, nullptr, 1);

    // 3) key smear
    smear_k<<<dim3((NH + 255) / 256), dim3(256), 0, stream>>>(kb, alpha, ksb);

    // 4) causal flash attention (32-query tiles)
    attn_kernel<<<dim3(S_ / 32, H_, B_), dim3(32), 0, stream>>>(qb, ksb, vtb, mhb);

    // 5) output projection (+bias) -> fp32 d_out
    gemm_bf16<<<ggrid, dim3(32), 0, stream>>>(mhb, wob, d_out, b_o, 2);
}
